// L2MLoRA_3805341024602
// MI455X (gfx1250) — matrix operand order repaired and checked by reading, not compiled, not run on hardware
//
#include <hip/hip_runtime.h>

// L2M LoRA forward for MI455X (gfx1250, wave32).
//   out(b,n,o) = sum_d x(b,n,d)*W(o,d) + bias(o) + 2 * sum_r T(b,n,r)*Bsel(r,o)
//   T = x @ A_sel (rank 8), precomputed by a tiny helper kernel into d_ws.
// Main GEMM: V_WMMA_F32_16X16X32_BF16, fp32 accumulation.
// Block tile 256(M) x 128(N), K-step 32; 8 wave32s in a 4x2 grid, each wave
// owns a 64x64 patch = 16 WMMA tiles fed by 4 A + 4 B fragments (4x reuse).

typedef __attribute__((ext_vector_type(16))) __bf16 bf16x16;
typedef __attribute__((ext_vector_type(8)))  __bf16 bf16x8;
typedef __attribute__((ext_vector_type(8)))  float  f32x8;

#define SHUF16(a, b) __builtin_shufflevector((a), (b), 0,1,2,3,4,5,6,7,8,9,10,11,12,13,14,15)

// ---------------------------------------------------------------------------
// Kernel 1: T[b,n,r] = sum_d x[b,n,d] * A_pool[sel(b), d, r]   (T -> d_ws)
// grid: 8 batches * 64 chunks of 32 rows; 256 threads = (32 n) x (8 r)
// ---------------------------------------------------------------------------
__global__ __launch_bounds__(256) void lora_xA_kernel(
    const float* __restrict__ x, const float* __restrict__ Apool,
    const long long* __restrict__ idx, float* __restrict__ T)
{
    __shared__ float As[1024 * 8];                 // A_sel, 32 KB
    const int bid = blockIdx.x;
    const int b   = bid >> 6;
    const int nb  = (bid & 63) * 32;
    const int sel = (int)idx[b];

    const float4* Ap4 = (const float4*)(Apool + (size_t)sel * 8192);
    float4*       As4 = (float4*)As;
    for (int j = threadIdx.x; j < 2048; j += 256) As4[j] = Ap4[j];
    __syncthreads();

    const int n = nb + (threadIdx.x >> 3);
    const int r = threadIdx.x & 7;
    const float* xr = x + (size_t)(b * 2048 + n) * 1024;

    float acc = 0.0f;
    for (int d = 0; d < 1024; d += 4) {
        const float4 xv = *(const float4*)(xr + d);
        acc += xv.x * As[(d + 0) * 8 + r];
        acc += xv.y * As[(d + 1) * 8 + r];
        acc += xv.z * As[(d + 2) * 8 + r];
        acc += xv.w * As[(d + 3) * 8 + r];
    }
    T[(size_t)(b * 2048 + n) * 8 + r] = acc;
}

// ---------------------------------------------------------------------------
// Kernel 2: fused GEMM + bias + LoRA epilogue.
// grid = dim3(8 /*n tiles*/, 64 /*m tiles of 256 rows*/); 256 threads.
// ---------------------------------------------------------------------------
__global__ __launch_bounds__(256) void lora_fused_gemm(
    const float* __restrict__ x, const float* __restrict__ weight,
    const float* __restrict__ bias, const float* __restrict__ Bpool,
    const long long* __restrict__ idx, const float* __restrict__ T,
    float* __restrict__ out)
{
    constexpr int KP = 40;                          // padded bf16 row stride (80 B)
    __shared__ __bf16 Xs[256 * KP];                 // x tile, bf16        (20 KB)
    __shared__ __bf16 Ws[128 * KP];                 // weight tile, bf16   (10 KB)
    __shared__ float  Ts[256 * 8];                  // LoRA T rows          (8 KB)
    __shared__ float  Bls[128 * 8];                 // B_sel[:, n0:n0+128], [n][r]
    __shared__ float  BiasS[128];

    const int tid = threadIdx.x;
    const int m0  = blockIdx.y * 256;               // global row (b*2048+n)
    const int n0  = blockIdx.x * 128;               // output column
    const int b   = m0 >> 11;                       // 2048 rows per batch
    const int sel = (int)idx[b];

    // -------- stage per-block LoRA/bias data (K-invariant) --------
    {
        const float4* Tsrc = (const float4*)(T + (size_t)m0 * 8);
        ((float4*)Ts)[tid]       = Tsrc[tid];
        ((float4*)Ts)[tid + 256] = Tsrc[tid + 256];
        for (int j = tid; j < 128 * 8; j += 256) {
            const int n = j >> 3, r = j & 7;
            Bls[j] = Bpool[(size_t)sel * 8192 + r * 1024 + n0 + n];
        }
        if (tid < 128) BiasS[tid] = bias[n0 + tid];
    }

    const int lane = tid & 31;
    const int w    = tid >> 5;
    const int half = lane >> 4;                     // K/M half select per ISA layout
    const int l16  = lane & 15;
    const int wm   = (w & 3) * 64;                  // wave M offset (4 m tiles)
    const int wn   = (w >> 2) * 64;                 // wave N offset (4 n tiles)

    f32x8 acc[4][4] = {};

    const int ldRow = tid >> 3;                     // 0..31
    const int ldK   = (tid & 7) * 4;                // 0..28

    for (int kk = 0; kk < 1024; kk += 32) {
        // -------- global fp32 -> bf16 LDS tiles --------
        #pragma unroll
        for (int p = 0; p < 8; ++p) {               // X: 256 rows
            const int row = p * 32 + ldRow;
            const float4 xv = *(const float4*)(x + (size_t)(m0 + row) * 1024 + kk + ldK);
            __bf16* px = &Xs[row * KP + ldK];
            px[0] = (__bf16)xv.x; px[1] = (__bf16)xv.y;
            px[2] = (__bf16)xv.z; px[3] = (__bf16)xv.w;
        }
        #pragma unroll
        for (int p = 0; p < 4; ++p) {               // W: 128 out-cols
            const int row = p * 32 + ldRow;
            const float4 wv = *(const float4*)(weight + (size_t)(n0 + row) * 1024 + kk + ldK);
            __bf16* pw = &Ws[row * KP + ldK];
            pw[0] = (__bf16)wv.x; pw[1] = (__bf16)wv.y;
            pw[2] = (__bf16)wv.z; pw[3] = (__bf16)wv.w;
        }
        if (kk + 32 < 1024) {                        // gfx1250 global_prefetch_b8
            __builtin_prefetch(x      + (size_t)(m0 + ldRow) * 1024 + kk + 32 + ldK, 0, 0);
            __builtin_prefetch(weight + (size_t)(n0 + ldRow) * 1024 + kk + 32 + ldK, 0, 0);
        }
        __syncthreads();

        // -------- fragments (contiguous K order: k = 16*half + s) + 16 WMMAs --------
        bf16x16 bfrag[4];
        #pragma unroll
        for (int nt = 0; nt < 4; ++nt) {
            const int c = wn + nt * 16 + l16;        // B: lane = column, K half by lane/16
            const bf16x8 blo = *(const bf16x8*)&Ws[c * KP + half * 16];
            const bf16x8 bhi = *(const bf16x8*)&Ws[c * KP + half * 16 + 8];
            bfrag[nt] = SHUF16(blo, bhi);
        }
        #pragma unroll
        for (int mt = 0; mt < 4; ++mt) {
            const int r = wm + mt * 16 + l16;        // A: lane = row, K 0-15 vs 16-31
            const bf16x8 alo = *(const bf16x8*)&Xs[r * KP + half * 16];
            const bf16x8 ahi = *(const bf16x8*)&Xs[r * KP + half * 16 + 8];
            const bf16x16 afrag = SHUF16(alo, ahi);
            #pragma unroll
            for (int nt = 0; nt < 4; ++nt) {
                acc[mt][nt] = __builtin_amdgcn_wmma_f32_16x16x32_bf16(
                    false, afrag, false, bfrag[nt], (short)0, acc[mt][nt], false, false);
            }
        }
        __syncthreads();
    }

    // -------- epilogue: bias + 2 * (T @ B_sel), then store --------
    float Bv[4][8], bv[4];
    #pragma unroll
    for (int nt = 0; nt < 4; ++nt) {
        const int n = wn + nt * 16 + l16;
        bv[nt] = BiasS[n];
        #pragma unroll
        for (int r2 = 0; r2 < 8; ++r2) Bv[nt][r2] = Bls[n * 8 + r2];
    }
    #pragma unroll
    for (int mt = 0; mt < 4; ++mt) {
        #pragma unroll
        for (int v = 0; v < 8; ++v) {
            const int ml = wm + mt * 16 + half * 8 + v;   // C/D layout: M = v (+8 for hi lanes)
            const float4 t0 = *(const float4*)&Ts[ml * 8];
            const float4 t1 = *(const float4*)&Ts[ml * 8 + 4];
            #pragma unroll
            for (int nt = 0; nt < 4; ++nt) {
                const float lora = t0.x * Bv[nt][0] + t0.y * Bv[nt][1]
                                 + t0.z * Bv[nt][2] + t0.w * Bv[nt][3]
                                 + t1.x * Bv[nt][4] + t1.y * Bv[nt][5]
                                 + t1.z * Bv[nt][6] + t1.w * Bv[nt][7];
                const int gn = n0 + wn + nt * 16 + l16;
                out[(size_t)(m0 + ml) * 1024 + gn] = acc[mt][nt][v] + bv[nt] + 2.0f * lora;
            }
        }
    }
}

// ---------------------------------------------------------------------------
extern "C" void kernel_launch(void* const* d_in, const int* in_sizes, int n_in,
                              void* d_out, int out_size, void* d_ws, size_t ws_size,
                              hipStream_t stream)
{
    (void)in_sizes; (void)n_in; (void)out_size; (void)ws_size;
    const float*     x      = (const float*)d_in[0];      // (8,2048,1024) f32
    const long long* idx    = (const long long*)d_in[1];  // (8,1) int64
    /* d_in[2] = frozen_mask: gradient-only, unused in forward */
    const float*     weight = (const float*)d_in[3];      // (1024,1024) f32
    const float*     bias   = (const float*)d_in[4];      // (1024,) f32
    const float*     Apool  = (const float*)d_in[5];      // (64,1024,8) f32
    const float*     Bpool  = (const float*)d_in[6];      // (64,8,1024) f32
    float*           out    = (float*)d_out;              // (8,2048,1024) f32
    float*           T      = (float*)d_ws;               // 8*2048*8 f32 = 512 KB

    lora_xA_kernel<<<512, 256, 0, stream>>>(x, Apool, idx, T);
    lora_fused_gemm<<<dim3(8, 64), 256, 0, stream>>>(x, weight, bias, Bpool, idx, T, out);
}
